// PhysicsInformedRegressor_9096740733455
// MI455X (gfx1250) — matrix operand order repaired and checked by reading, not compiled, not run on hardware
//
#include <hip/hip_runtime.h>
#include <hip/hip_bf16.h>

typedef __attribute__((ext_vector_type(16))) _Float16 v16h;
typedef __attribute__((ext_vector_type(8)))  float    v8f;
typedef __attribute__((ext_vector_type(4)))  unsigned uint4v;

#define NB        65536      // batch
#define HID       256        // hidden width
#define NCH       7          // value + 3 grads + 3 hess-diag channels
#define SPW       16         // samples per workgroup

// LDS layout (byte offsets)
#define LDSW_OFF   0                       // 256*256 f16 weights (row=n, swizzled k) = 131072 B
#define ACT0_OFF   131072                  // 7*16*256 f16 activations               =  57344 B
#define ACT1_OFF   188416                  // double buffer                          =  57344 B
#define XT_OFF     245760                  // 16*3 f32 sample coords                 =    192 B
#define WOUT_OFF   245952                  // 256 f32 output weights                 =   1024 B
#define SMEM_BYTES 246976

#if __has_builtin(__builtin_amdgcn_tanhf)
#define FAST_TANH(x) __builtin_amdgcn_tanhf(x)
#elif __has_builtin(__builtin_amdgcn_tanh_f32)
#define FAST_TANH(x) __builtin_amdgcn_tanh_f32(x)
#else
#define FAST_TANH(x) tanhf(x)
#endif

union Frag { v16h h; uint4v q[2]; unsigned u[8]; };

// XOR-swizzled element index: row-major [row][k] with K grouped in 16-element (32B)
// chunks, chunk index XORed with (row&15). Keeps 16B fragment loads contiguous while
// spreading lanes across LDS bank groups.
__device__ __forceinline__ int swz(int row, int k) {
  return (row << 8) + ((((k >> 4) ^ row) & 15) << 4) + (k & 15);
}

// Load the 8 A-fragments (all K) for channel c. 16-bit A 16x32 layout:
// lane&15 = M, half = lane>>4; u[0..3] = K {8*half+0..7}, u[4..7] = K {16+8*half+0..7}.
__device__ __forceinline__ void load_afrags(Frag* af, const _Float16* __restrict__ act,
                                            int c, int M, int half) {
  const int row = c * SPW + M;
  #pragma unroll
  for (int kk = 0; kk < 8; ++kk) {
    #pragma unroll
    for (int sub = 0; sub < 2; ++sub) {
      const int e = (row << 8) + ((((2 * kk + sub) ^ row) & 15) << 4) + (half << 3);
      af[kk].q[sub] = *(const uint4v*)__builtin_assume_aligned((const void*)(act + e), 16);
    }
  }
}

// Load the 8 B-fragments for output column j. 16-bit B 32x16 layout:
// lane&15 = N, half = lane>>4; u[0..3] = K {8*half+0..7}, u[4..7] = K {16+8*half+0..7}.
__device__ __forceinline__ void load_bfrags(Frag* bf, const _Float16* __restrict__ w,
                                            int j, int half) {
  #pragma unroll
  for (int kk = 0; kk < 8; ++kk) {
    const int e0 = (j << 8) + ((((2 * kk) ^ j) & 15) << 4) + (half << 3);
    const int e1 = (j << 8) + ((((2 * kk + 1) ^ j) & 15) << 4) + (half << 3);
    bf[kk].q[0] = *(const uint4v*)__builtin_assume_aligned((const void*)(w + e0), 16);
    bf[kk].q[1] = *(const uint4v*)__builtin_assume_aligned((const void*)(w + e1), 16);
  }
}

// One-time prep: convert W1..W3 (f32 row-major [k][n]) into f16 images in d_ws laid out
// exactly as the LDS weight buffer (row = n, XOR-swizzled K chunks). Main-kernel staging
// then becomes a straight 128KB coalesced copy.
__global__ void __launch_bounds__(256)
prep_weights_kernel(const float* __restrict__ W1, const float* __restrict__ W2,
                    const float* __restrict__ W3, _Float16* __restrict__ wsw) {
  const int e = blockIdx.x * 256 + threadIdx.x;   // dest element within one layer image
  const int l = blockIdx.y;
  const float* W = (l == 0) ? W1 : ((l == 1) ? W2 : W3);
  const int n  = e >> 8;
  const int cp = (e >> 4) & 15;
  const int k  = (((cp ^ n) & 15) << 4) | (e & 15);
  wsw[l * (HID * HID) + e] = (_Float16)W[k * HID + n];
}

__global__ void __launch_bounds__(256)
pinn_fused_kernel(const float* __restrict__ x,
                  const float* __restrict__ W0, const float* __restrict__ b0,
                  const float* __restrict__ b1, const float* __restrict__ b2,
                  const float* __restrict__ b3,
                  const _Float16* __restrict__ wsw,
                  const float* __restrict__ Wout, const float* __restrict__ bout,
                  float* __restrict__ out) {
  extern __shared__ char smem[];
  _Float16* ldsw = (_Float16*)(smem + LDSW_OFF);
  float* xt = (float*)(smem + XT_OFF);
  float* wo = (float*)(smem + WOUT_OFF);

  const int tid  = threadIdx.x;
  const int lane = tid & 31;
  const int wave = tid >> 5;
  const int wg   = blockIdx.x;
  const int M    = lane & 15;
  const int half = lane >> 4;

  if (tid < SPW * 3) xt[tid] = x[wg * (SPW * 3) + tid];
  wo[tid] = Wout[tid];
  __syncthreads();

  // ---- Layer 0 (3 -> 256): seed all 7 channels (VALU; tiny) ----
  {
    const int j = tid;
    const float w0 = W0[j], w1 = W0[HID + j], w2 = W0[2 * HID + j];
    const float bj = b0[j];
    _Float16* A = (_Float16*)(smem + ACT0_OFF);
    #pragma unroll 4
    for (int s = 0; s < SPW; ++s) {
      float z  = xt[s*3]*w0 + xt[s*3+1]*w1 + xt[s*3+2]*w2 + bj;
      float t  = FAST_TANH(z);
      float tp = 1.f - t * t;
      float tpp = -2.f * t * tp;
      A[swz(0*SPW + s, j)] = (_Float16)t;
      A[swz(1*SPW + s, j)] = (_Float16)(tp * w0);
      A[swz(2*SPW + s, j)] = (_Float16)(tp * w1);
      A[swz(3*SPW + s, j)] = (_Float16)(tp * w2);
      A[swz(4*SPW + s, j)] = (_Float16)(tpp * w0 * w0);
      A[swz(5*SPW + s, j)] = (_Float16)(tpp * w1 * w1);
      A[swz(6*SPW + s, j)] = (_Float16)(tpp * w2 * w2);
    }
  }

  int cur = 0;

  for (int layer = 0; layer < 3; ++layer) {
    __syncthreads();               // prior acts written; prior weight reads done
    // Stage this layer's pre-swizzled f16 weight image: pure 128KB coalesced copy.
    const _Float16* Wf16 = wsw + layer * (HID * HID);
    const float* b = (layer == 0) ? b1 : ((layer == 1) ? b2 : b3);
    for (int i = tid; i < (HID * HID) / 8; i += 256) {
      const int e = i << 3;
      *(uint4v*)__builtin_assume_aligned((void*)(ldsw + e), 16) =
          *(const uint4v*)__builtin_assume_aligned((const void*)(Wf16 + e), 16);
    }
    __syncthreads();

    const _Float16* A    = (const _Float16*)(smem + (cur ? ACT1_OFF : ACT0_OFF));
    _Float16*       Anew = (_Float16*)(smem + (cur ? ACT0_OFF : ACT1_OFF));

    // This wave owns columns [32*wave, 32*wave+32): two 16-wide N-tiles.
    const int j0 = (wave << 5) + M;
    const int j1 = j0 + 16;

    Frag bf0[8], bf1[8];
    load_bfrags(bf0, ldsw, j0, half);
    load_bfrags(bf1, ldsw, j1, half);
    const float bj0 = b[j0];
    const float bj1 = b[j1];

    Frag af[8];
    float tpv0[8], tppv0[8], tpv1[8], tppv1[8];

    // ---- value channel: both tiles share the A operand ----
    {
      load_afrags(af, A, 0, M, half);
      v8f z0 = {}, z1 = {};
      #pragma unroll
      for (int kk = 0; kk < 8; ++kk) {
        z0 = __builtin_amdgcn_wmma_f32_16x16x32_f16(false, af[kk].h, false, bf0[kk].h,
                                                    (short)0, z0, false, false);
        z1 = __builtin_amdgcn_wmma_f32_16x16x32_f16(false, af[kk].h, false, bf1[kk].h,
                                                    (short)0, z1, false, false);
      }
      #pragma unroll
      for (int r = 0; r < 8; ++r) {
        const int s = r + (half << 3);       // C/D: VGPR r -> row M = r + 8*half
        float t  = FAST_TANH(z0[r] + bj0);
        tpv0[r]  = 1.f - t * t;
        tppv0[r] = -2.f * t * tpv0[r];
        Anew[swz(0*SPW + s, j0)] = (_Float16)t;
        float u  = FAST_TANH(z1[r] + bj1);
        tpv1[r]  = 1.f - u * u;
        tppv1[r] = -2.f * u * tpv1[r];
        Anew[swz(0*SPW + s, j1)] = (_Float16)u;
      }
    }

    // ---- (gradient, hessian) channel pairs: keep zg live only per pair ----
    #pragma unroll
    for (int c = 1; c <= 3; ++c) {
      load_afrags(af, A, c, M, half);
      v8f g0 = {}, g1 = {};
      #pragma unroll
      for (int kk = 0; kk < 8; ++kk) {
        g0 = __builtin_amdgcn_wmma_f32_16x16x32_f16(false, af[kk].h, false, bf0[kk].h,
                                                    (short)0, g0, false, false);
        g1 = __builtin_amdgcn_wmma_f32_16x16x32_f16(false, af[kk].h, false, bf1[kk].h,
                                                    (short)0, g1, false, false);
      }
      load_afrags(af, A, c + 3, M, half);
      v8f h0 = {}, h1 = {};
      #pragma unroll
      for (int kk = 0; kk < 8; ++kk) {
        h0 = __builtin_amdgcn_wmma_f32_16x16x32_f16(false, af[kk].h, false, bf0[kk].h,
                                                    (short)0, h0, false, false);
        h1 = __builtin_amdgcn_wmma_f32_16x16x32_f16(false, af[kk].h, false, bf1[kk].h,
                                                    (short)0, h1, false, false);
      }
      #pragma unroll
      for (int r = 0; r < 8; ++r) {
        const int s = r + (half << 3);
        Anew[swz(c*SPW + s, j0)]     = (_Float16)(tpv0[r] * g0[r]);
        Anew[swz((c+3)*SPW + s, j0)] = (_Float16)(tppv0[r]*g0[r]*g0[r] + tpv0[r]*h0[r]);
        Anew[swz(c*SPW + s, j1)]     = (_Float16)(tpv1[r] * g1[r]);
        Anew[swz((c+3)*SPW + s, j1)] = (_Float16)(tppv1[r]*g1[r]*g1[r] + tpv1[r]*h1[r]);
      }
    }
    cur ^= 1;
  }

  __syncthreads();
  // ---- Output layer: 7x16 length-256 dot products with Wout (b128 reads) ----
  if (tid < NCH * SPW) {
    const int c = tid >> 4;
    const int s = tid & 15;
    const _Float16* A = (const _Float16*)(smem + (cur ? ACT1_OFF : ACT0_OFF));
    const int row = c * SPW + s;
    float sum = (c == 0) ? bout[0] : 0.f;
    for (int cp = 0; cp < 16; ++cp) {
      const int ebase = (row << 8) + (cp << 4);
      const int kbase = ((cp ^ row) & 15) << 4;
      union { uint4v q; _Float16 h[8]; } lo, hi;
      lo.q = *(const uint4v*)__builtin_assume_aligned((const void*)(A + ebase), 16);
      hi.q = *(const uint4v*)__builtin_assume_aligned((const void*)(A + ebase + 8), 16);
      #pragma unroll
      for (int i = 0; i < 8; ++i) {
        sum += (float)lo.h[i] * wo[kbase + i];
        sum += (float)hi.h[i] * wo[kbase + 8 + i];
      }
    }
    // reference packs [u, grad(3), hess_diag(3)] -> channel index == output column
    out[(wg * SPW + s) * NCH + c] = sum;
  }
}

extern "C" void kernel_launch(void* const* d_in, const int* in_sizes, int n_in,
                              void* d_out, int out_size, void* d_ws, size_t ws_size,
                              hipStream_t stream) {
  const float* x    = (const float*)d_in[0];
  const float* W0   = (const float*)d_in[1];
  const float* b0   = (const float*)d_in[2];
  const float* W1   = (const float*)d_in[3];
  const float* b1   = (const float*)d_in[4];
  const float* W2   = (const float*)d_in[5];
  const float* b2   = (const float*)d_in[6];
  const float* W3   = (const float*)d_in[7];
  const float* b3   = (const float*)d_in[8];
  const float* Wout = (const float*)d_in[9];
  const float* bout = (const float*)d_in[10];
  float* out = (float*)d_out;
  _Float16* wsw = (_Float16*)d_ws;   // 3 * 256*256 f16 = 384KB swizzled weight images

  (void)hipFuncSetAttribute((const void*)pinn_fused_kernel,
                            hipFuncAttributeMaxDynamicSharedMemorySize, SMEM_BYTES);

  // One pass: build swizzled f16 weight images in workspace (768KB read, L2-resident).
  hipLaunchKernelGGL(prep_weights_kernel, dim3((HID * HID) / 256, 3), dim3(256), 0,
                     stream, W1, W2, W3, wsw);

  dim3 grid(NB / SPW);   // 4096 workgroups x 16 samples
  dim3 block(256);       // 8 wave32
  hipLaunchKernelGGL(pinn_fused_kernel, grid, block, SMEM_BYTES, stream,
                     x, W0, b0, b1, b2, b3, wsw, Wout, bout, out);
}
